// GraphEncoder_12627203850709
// MI455X (gfx1250) — matrix operand order repaired and checked by reading, not compiled, not run on hardware
//
#include <hip/hip_runtime.h>
#include <hip/hip_bf16.h>

// ---------------------------------------------------------------------------
// Types for WMMA fragments (wave32, gfx1250)
// ---------------------------------------------------------------------------
typedef __attribute__((ext_vector_type(16))) __bf16        v16bf;
typedef __attribute__((ext_vector_type(8)))  float         v8f;
typedef __attribute__((ext_vector_type(4)))  unsigned int  u32x4;

union FragBF { v16bf v; u32x4 q[2]; };

// fp32 -> bf16 round-to-nearest-even
__device__ __forceinline__ unsigned short f2bf(float f) {
    unsigned int u = __float_as_uint(f);
    unsigned int r = u + 0x7FFFu + ((u >> 16) & 1u);
    return (unsigned short)(r >> 16);
}

// order-preserving float<->uint mapping (for atomicMax-based segment max)
__device__ __forceinline__ unsigned int ordf(float f) {
    unsigned int u = __float_as_uint(f);
    return ((int)u >= 0) ? (u ^ 0x80000000u) : ~u;
}
__device__ __forceinline__ float unordf(unsigned int o) {
    unsigned int u = (o & 0x80000000u) ? (o ^ 0x80000000u) : ~o;
    return __uint_as_float(u);
}

// edge k -> (src, dst), faithful to the reference's stack+reshape semantics
__device__ __forceinline__ void edge_pair(const long long* __restrict__ adj,
                                          long long k, long long E, long long BE,
                                          long long& s, long long& d) {
    if (k < BE) {
        long long r  = (k / E) & 1;
        long long e  = k % E;
        s = adj[r * E + e];
        long long j  = BE + k;
        long long r2 = (j / E) & 1;
        d = adj[r2 * E + (j % E)];
    } else {                    // appended self loops
        s = d = k - BE;
    }
}

// ---------------------------------------------------------------------------
// GEMM: C[M,N] = A[M,K](bf16, row-major) * Bt[N,K](bf16, pre-transposed) (f32 out)
// One wave -> one 16x16 C tile, K-loop in steps of 32 via v_wmma_f32_16x16x32_bf16
// ---------------------------------------------------------------------------
template <int K, int N>
__global__ void gemm_bf16_wmma(const unsigned short* __restrict__ A,
                               const unsigned short* __restrict__ Bt,
                               float* __restrict__ C, int M) {
    const int lane  = threadIdx.x & 31;
    const int wave  = threadIdx.x >> 5;
    const int tm    = blockIdx.x;                 // M/16 tiles
    const int tn    = blockIdx.y * 8 + wave;      // N/16 tiles (8 waves per block)
    if (tn * 16 >= N) return;
    const int  c16 = lane & 15;
    const bool hi  = lane >= 16;

    const unsigned short* Arow = A  + (size_t)(tm * 16 + c16) * K + (hi ? 8 : 0);
    const unsigned short* Brow = Bt + (size_t)(tn * 16 + c16) * K + (hi ? 8 : 0);

    v8f acc = {};
#pragma unroll
    for (int kb = 0; kb < K; kb += 32) {
        FragBF a, b;
        a.q[0] = *(const u32x4*)(Arow + kb);        // K = kb+off .. +7
        a.q[1] = *(const u32x4*)(Arow + kb + 16);   // K = kb+off+16 .. +23
        b.q[0] = *(const u32x4*)(Brow + kb);        // K = kb+off .. +7
        b.q[1] = *(const u32x4*)(Brow + kb + 16);   // K = kb+off+16 .. +23
        acc = __builtin_amdgcn_wmma_f32_16x16x32_bf16(
                  false, a.v, false, b.v, (short)0, acc, false, false);
    }

    float* Cp = C + (size_t)(tm * 16 + (hi ? 8 : 0)) * N + tn * 16 + c16;
#pragma unroll
    for (int i = 0; i < 8; ++i) Cp[(size_t)i * N] = acc[i];
}

// ---------------------------------------------------------------------------
// Small prep / epilogue kernels
// ---------------------------------------------------------------------------
__global__ void k_tobf(const float* __restrict__ x, unsigned short* __restrict__ y,
                       long long n) {
    long long i = blockIdx.x * (long long)blockDim.x + threadIdx.x;
    if (i < n) y[i] = f2bf(x[i]);
}

// Wt[n][k] = bf16(W[k][n])
__global__ void k_transpose_bf(const float* __restrict__ W, unsigned short* __restrict__ Wt,
                               int K, int N) {
    int i = blockIdx.x * blockDim.x + threadIdx.x;
    if (i >= K * N) return;
    int n = i / K, k = i % K;
    Wt[i] = f2bf(W[(size_t)k * N + n]);
}

__global__ void k_bias_relu_bf(const float* __restrict__ x, const float* __restrict__ b,
                               unsigned short* __restrict__ y, int HD, long long tot) {
    long long i = blockIdx.x * (long long)blockDim.x + threadIdx.x;
    if (i >= tot) return;
    float v = x[i] + b[(int)(i % HD)];
    y[i] = f2bf(v > 0.f ? v : 0.f);
}

__global__ void k_fc_out(const float* __restrict__ C, const float* __restrict__ b,
                         float* __restrict__ y, int HD, long long tot) {
    long long i = blockIdx.x * (long long)blockDim.x + threadIdx.x;
    if (i >= tot) return;
    float v = C[i] + b[(int)(i % HD)];
    y[i] = v > 0.f ? v : 0.f;
}

// adj_rep flat copy: flat[j] = adj[(j/E)&1][j%E], j in [0, 2*B*E)
__global__ void k_adj_rep(const long long* __restrict__ adj, long long* __restrict__ out,
                          long long E, long long tot) {
    long long j = blockIdx.x * (long long)blockDim.x + threadIdx.x;
    if (j >= tot) return;
    long long r = (j / E) & 1;
    out[j] = adj[r * E + (j % E)];
}

// ---------------------------------------------------------------------------
// Attention scores: a_s[n,h] = <h_lin[n, h*D:(h+1)*D], att_s[h]>, wave per (n,h)
// ---------------------------------------------------------------------------
template <int H, int D>
__global__ void k_att_scores(const float* __restrict__ hlin,
                             const float* __restrict__ att_s, const float* __restrict__ att_d,
                             float* __restrict__ a_s, float* __restrict__ a_d, long long NN) {
    long long gw = blockIdx.x * (long long)(blockDim.x >> 5) + (threadIdx.x >> 5);
    if (gw >= NN * H) return;
    int lane = threadIdx.x & 31;
    int h = (int)(gw % H);
    long long n = gw / H;
    const float* row = hlin + n * (long long)(H * D) + h * D;
    float ss = 0.f, sd = 0.f;
#pragma unroll
    for (int i = lane; i < D; i += 32) {
        float v = row[i];
        ss += v * att_s[h * D + i];
        sd += v * att_d[h * D + i];
    }
#pragma unroll
    for (int off = 16; off; off >>= 1) {
        ss += __shfl_xor(ss, off, 32);
        sd += __shfl_xor(sd, off, 32);
    }
    if (lane == 0) { a_s[gw] = ss; a_d[gw] = sd; }
}

// ---------------------------------------------------------------------------
// Edge pass 1: e = leaky_relu(a_s[src]+a_d[dst]); segment max via ordered atomicMax
// ---------------------------------------------------------------------------
template <int H>
__global__ void k_edge_max(const long long* __restrict__ adj, long long E, long long BE,
                           long long NN, const float* __restrict__ a_s,
                           const float* __restrict__ a_d, unsigned int* __restrict__ m_ord) {
    long long t = blockIdx.x * (long long)blockDim.x + threadIdx.x;
    if (t >= (BE + NN) * H) return;
    int h = (int)(t % H);
    long long k = t / H, s, d;
    edge_pair(adj, k, E, BE, s, d);
    float v = a_s[s * H + h] + a_d[d * H + h];
    v = v > 0.f ? v : 0.2f * v;
    atomicMax(&m_ord[d * H + h], ordf(v));
}

// Edge pass 2: ex = exp(e - m[dst]); den[dst] += ex (hw f32 atomic)
template <int H>
__global__ void k_edge_expsum(const long long* __restrict__ adj, long long E, long long BE,
                              long long NN, const float* __restrict__ a_s,
                              const float* __restrict__ a_d,
                              const unsigned int* __restrict__ m_ord,
                              float* __restrict__ exbuf, float* __restrict__ den) {
    long long t = blockIdx.x * (long long)blockDim.x + threadIdx.x;
    if (t >= (BE + NN) * H) return;
    int h = (int)(t % H);
    long long k = t / H, s, d;
    edge_pair(adj, k, E, BE, s, d);
    float v = a_s[s * H + h] + a_d[d * H + h];
    v = v > 0.f ? v : 0.2f * v;
    float ex = __expf(v - unordf(m_ord[d * H + h]));
    exbuf[k * H + h] = ex;
    unsafeAtomicAdd(&den[d * H + h], ex);
}

// Edge pass 3: out[dst] += alpha * h[src]; one wave per edge, HD/32 channels per lane
template <int H, int D>
__global__ void k_edge_agg(const long long* __restrict__ adj, long long E, long long BE,
                           long long NN, const float* __restrict__ exbuf,
                           const float* __restrict__ den, const float* __restrict__ hsrc,
                           float* __restrict__ out) {
    constexpr int HD = H * D, PER = HD / 32;
    long long k = blockIdx.x * (long long)(blockDim.x >> 5) + (threadIdx.x >> 5);
    if (k >= BE + NN) return;
    int lane = threadIdx.x & 31;
    long long s, d;
    edge_pair(adj, k, E, BE, s, d);
    int c0 = lane * PER;
    int h  = c0 / D;                       // constant within a lane (D multiple of PER)
    float alpha = exbuf[k * H + h] / (den[d * H + h] + 1e-16f);
    const float* sp = hsrc + s * (long long)HD + c0;
    float*       dp = out  + d * (long long)HD + c0;
#pragma unroll
    for (int i = 0; i < PER; ++i)
        unsafeAtomicAdd(&dp[i], sp[i] * alpha);
}

// ---------------------------------------------------------------------------
// Host orchestration
// ---------------------------------------------------------------------------
extern "C" void kernel_launch(void* const* d_in, const int* in_sizes, int n_in,
                              void* d_out, int out_size, void* d_ws, size_t ws_size,
                              hipStream_t stream) {
    (void)n_in; (void)out_size; (void)ws_size;
    const float* x     = (const float*)d_in[0];
    const float* W1    = (const float*)d_in[1];
    const float* as1   = (const float*)d_in[2];
    const float* ad1   = (const float*)d_in[3];
    const float* b1    = (const float*)d_in[4];
    const float* W2    = (const float*)d_in[5];
    const float* as2   = (const float*)d_in[6];
    const float* ad2   = (const float*)d_in[7];
    const float* b2    = (const float*)d_in[8];
    const float* Wfc   = (const float*)d_in[9];
    const float* bfc   = (const float*)d_in[10];
    const long long* adj = (const long long*)d_in[11];

    const int  F  = 256;
    const long long NN = in_sizes[0] / F;        // 16384 nodes (B*N)
    const long long E  = in_sizes[11] / 2;       // 65536
    const long long BE = 8 * E;                  // B = 8 replications
    const long long ET = BE + NN;                // edges incl. self loops

    // --- workspace carve-up ---
    char* w = (char*)d_ws;
    size_t off = 0;
    auto carve = [&](size_t bytes) { void* p = w + off; off = (off + bytes + 255) & ~(size_t)255; return p; };
    unsigned short* Abf  = (unsigned short*)carve((size_t)NN * 256 * 2);  // bf16 activations
    unsigned short* Wt1  = (unsigned short*)carve(256 * 256 * 2);
    unsigned short* Wt2  = (unsigned short*)carve(256 * 128 * 2);
    unsigned short* Wtfc = (unsigned short*)carve(128 * 128 * 2);
    float* Hlin = (float*)carve((size_t)NN * 256 * 4);                    // GEMM out / h_lin
    float* Agg  = (float*)carve((size_t)NN * 256 * 4);                    // aggregation
    float* aS   = (float*)carve((size_t)NN * 2 * 4);
    float* aD   = (float*)carve((size_t)NN * 2 * 4);
    unsigned int* Mord = (unsigned int*)carve((size_t)NN * 2 * 4);
    float* Den  = (float*)carve((size_t)NN * 2 * 4);
    float* Ex   = (float*)carve((size_t)ET * 2 * 4);

    float*      y_out   = (float*)d_out;                       // [NN,128]
    long long*  adj_out = (long long*)(y_out + (size_t)NN * 128);

    const int TPB = 256;
    auto blks = [](long long n, int t) { return (unsigned)((n + t - 1) / t); };

    // --- weight prep (tiny) ---
    k_transpose_bf<<<blks(256 * 256, TPB), TPB, 0, stream>>>(W1, Wt1, 256, 256);
    k_transpose_bf<<<blks(256 * 128, TPB), TPB, 0, stream>>>(W2, Wt2, 256, 128);
    k_transpose_bf<<<blks(128 * 128, TPB), TPB, 0, stream>>>(Wfc, Wtfc, 128, 128);
    k_tobf<<<blks(NN * 256, TPB), TPB, 0, stream>>>(x, Abf, NN * 256);

    // ======================= Layer 1 (H=2, D=128) =======================
    gemm_bf16_wmma<256, 256><<<dim3((unsigned)(NN / 16), 2), TPB, 0, stream>>>(Abf, Wt1, Hlin, (int)NN);
    k_att_scores<2, 128><<<blks(NN * 2 * 32, TPB), TPB, 0, stream>>>(Hlin, as1, ad1, aS, aD, NN);

    hipMemsetAsync(Mord, 0, (size_t)NN * 2 * 4, stream);
    hipMemsetAsync(Den,  0, (size_t)NN * 2 * 4, stream);
    hipMemsetAsync(Agg,  0, (size_t)NN * 256 * 4, stream);

    k_edge_max<2>   <<<blks(ET * 2, TPB), TPB, 0, stream>>>(adj, E, BE, NN, aS, aD, Mord);
    k_edge_expsum<2><<<blks(ET * 2, TPB), TPB, 0, stream>>>(adj, E, BE, NN, aS, aD, Mord, Ex, Den);
    k_edge_agg<2, 128><<<blks(ET * 32, TPB), TPB, 0, stream>>>(adj, E, BE, NN, Ex, Den, Hlin, Agg);

    k_bias_relu_bf<<<blks(NN * 256, TPB), TPB, 0, stream>>>(Agg, b1, Abf, 256, NN * 256);

    // ======================= Layer 2 (H=1, D=128) =======================
    gemm_bf16_wmma<256, 128><<<dim3((unsigned)(NN / 16), 1), TPB, 0, stream>>>(Abf, Wt2, Hlin, (int)NN);
    k_att_scores<1, 128><<<blks(NN * 1 * 32, TPB), TPB, 0, stream>>>(Hlin, as2, ad2, aS, aD, NN);

    hipMemsetAsync(Mord, 0, (size_t)NN * 1 * 4, stream);
    hipMemsetAsync(Den,  0, (size_t)NN * 1 * 4, stream);
    hipMemsetAsync(Agg,  0, (size_t)NN * 128 * 4, stream);

    k_edge_max<1>   <<<blks(ET * 1, TPB), TPB, 0, stream>>>(adj, E, BE, NN, aS, aD, Mord);
    k_edge_expsum<1><<<blks(ET * 1, TPB), TPB, 0, stream>>>(adj, E, BE, NN, aS, aD, Mord, Ex, Den);
    k_edge_agg<1, 128><<<blks(ET * 32, TPB), TPB, 0, stream>>>(adj, E, BE, NN, Ex, Den, Hlin, Agg);

    k_bias_relu_bf<<<blks(NN * 128, TPB), TPB, 0, stream>>>(Agg, b2, Abf, 128, NN * 128);

    // ======================= FC + outputs =======================
    gemm_bf16_wmma<128, 128><<<dim3((unsigned)(NN / 16), 1), TPB, 0, stream>>>(Abf, Wtfc, Hlin, (int)NN);
    k_fc_out<<<blks(NN * 128, TPB), TPB, 0, stream>>>(Hlin, bfc, y_out, 128, NN * 128);
    k_adj_rep<<<blks(2 * BE, TPB), TPB, 0, stream>>>(adj, adj_out, E, 2 * BE);
}
